// DKVMN_58944131170322
// MI455X (gfx1250) — matrix operand order repaired and checked by reading, not compiled, not run on hardware
//
#include <hip/hip_runtime.h>
#include <hip/hip_bf16.h>
#include <math.h>

typedef __attribute__((ext_vector_type(16))) _Float16 v16h;
typedef __attribute__((ext_vector_type(8)))  _Float16 v8h;
typedef __attribute__((ext_vector_type(8)))  float    v8f;
typedef __attribute__((ext_vector_type(4)))  int      v4i_;

#define NUM_C   100
#define DD      64
#define M_DIM   50
#define B_DIM   32
#define L_DIM   500
#define N_NODES 101
#define GHD     8

#if defined(__AMDGCN__) && __has_builtin(__builtin_amdgcn_global_load_async_to_lds_b128) && __has_builtin(__builtin_amdgcn_s_wait_asynccnt)
#define USE_ASYNC_LDS 1
#else
#define USE_ASYNC_LDS 0
#endif

// Fast reciprocal: v_rcp_f32 instead of IEEE divide expansion.
__device__ __forceinline__ float fastrcp_(float x){
#if defined(__AMDGCN__) && __has_builtin(__builtin_amdgcn_rcpf)
  return __builtin_amdgcn_rcpf(x);
#else
  return 1.0f/x;
#endif
}

// CDNA5 has a hardware TANH transcendental (TRANS32): single v_tanh_f32.
__device__ __forceinline__ float fasttanh_(float x){
#if defined(__AMDGCN__) && __has_builtin(__builtin_amdgcn_tanhf)
  return __builtin_amdgcn_tanhf(x);
#else
  return tanhf(x);
#endif
}

__device__ __forceinline__ float sigmoidf_(float x){
  return fastrcp_(1.0f + __expf(-x));
}

__device__ __forceinline__ v8f wmma_f16(v16h a, v16h b, v8f c){
  return __builtin_amdgcn_wmma_f32_16x16x32_f16(false, a, false, b, (short)0, c, false, false);
}

// A fragment straight from global memory (row-major [row][64] halves).
// ISA 7.12.2 16-bit A 16x32: lane<16 -> M=lane, halves0..7 = K ko..ko+7,
// halves8..15 = K ko+16..ko+23; lanes>=16 shift K by +8.  Two 16B loads.
__device__ __forceinline__ v16h load_a_g(const _Float16* __restrict__ g, int row0, int lane, int ko){
  int base = (row0 + (lane & 15))*DD + (lane >> 4)*8 + ko;
  v8h lo = *(const v8h*)(g + base);
  v8h hi = *(const v8h*)(g + base + 16);
  return __builtin_shufflevector(lo, hi, 0,1,2,3,4,5,6,7,8,9,10,11,12,13,14,15);
}

// B fragment from a pre-swizzled buffer: [(tile)*32 + lane]*16 contiguous halves.
__device__ __forceinline__ v16h load_b_sw(const _Float16* __restrict__ sw, int tile, int lane){
  const _Float16* p = sw + ((tile*32 + lane) << 4);
  v8h lo = *(const v8h*)(p);
  v8h hi = *(const v8h*)(p + 8);
  return __builtin_shufflevector(lo, hi, 0,1,2,3,4,5,6,7,8,9,10,11,12,13,14,15);
}

// Stage nhalves f16 from global to LDS (async path when available).
__device__ __forceinline__ void stage_lds(_Float16* dst, const _Float16* src,
                                          int nhalves, int tid, int nthr){
#if USE_ASYNC_LDS
  for (int i = tid*8; i < nhalves; i += nthr*8){
    __attribute__((address_space(1))) void* gvoid =
        (__attribute__((address_space(1))) void*)((void*)(src + i));
    __attribute__((address_space(3))) void* lvoid =
        (__attribute__((address_space(3))) void*)((void*)(dst + i));
    __builtin_amdgcn_global_load_async_to_lds_b128(
        (__attribute__((address_space(1))) v4i_*)gvoid,
        (__attribute__((address_space(3))) v4i_*)lvoid,
        0, 0);
  }
  __builtin_amdgcn_s_wait_asynccnt(0);
#else
  for (int i = tid; i < (nhalves >> 3); i += nthr)
    ((uint4*)dst)[i] = ((const uint4*)src)[i];
#endif
}

// ---------------------------------------------------------------- weight swizzle (once)
// Fragment order: entry = ((nt*KT + kt)*32 + lane)*16 + h
//   K = kt*32 + (lane>>4)*8 + (h&7) + (h>>3)*16 ; col = nt*16 + (lane&15)
__global__ void __launch_bounds__(256)
swizzle_weights_kernel(const float* __restrict__ Mk, const float* __restrict__ We,
                       const float* __restrict__ Wa, const float* __restrict__ Wf,
                       _Float16* __restrict__ swMk, _Float16* __restrict__ swWe,
                       _Float16* __restrict__ swWa, _Float16* __restrict__ swWf)
{
  int e = blockIdx.x*256 + threadIdx.x;
  if (e < 4096){                       // 64x64 weights, KT=2
    int h=e&15, lane=(e>>4)&31, kt=(e>>9)&1, nt=(e>>10)&3;
    int n=lane&15, hi=lane>>4;
    int K   = kt*32 + hi*8 + (h&7) + ((h>>3)<<4);
    int col = nt*16 + n;
    swMk[e] = (col < M_DIM) ? (_Float16)Mk[col*DD + K] : (_Float16)0.f; // B[K][m]=Mk[m][K]
    swWe[e] = (_Float16)We[K*DD + col];
    swWa[e] = (_Float16)Wa[K*DD + col];
  }
  if (e < 8192){                       // 128x64 weight, KT=4
    int h=e&15, lane=(e>>4)&31, kt=(e>>9)&3, nt=(e>>11)&3;
    int n=lane&15, hi=lane>>4;
    int K   = kt*32 + hi*8 + (h&7) + ((h>>3)<<4);
    int col = nt*16 + n;
    swWf[e] = (_Float16)Wf[K*DD + col];
  }
}

// ---------------------------------------------------------------- GCN (per-batch graph)
__global__ void gcn_kernel(const int* __restrict__ skill, const float* __restrict__ k_emb,
                           const float* __restrict__ Wg1, const float* __restrict__ bg1,
                           const float* __restrict__ Wg2, const float* __restrict__ bg2,
                           float* __restrict__ stu)
{
  __shared__ int   deg[N_NODES];
  __shared__ float dinv[N_NODES];
  __shared__ float h1[N_NODES*GHD];
  __shared__ float acc1[N_NODES*GHD];
  __shared__ float gact[N_NODES*GHD];
  __shared__ float gw[N_NODES*DD];
  __shared__ float acc2[N_NODES*DD];
  const int b = blockIdx.x, tid = threadIdx.x, nt = blockDim.x;
  const int* sk = skill + b*L_DIM;

  for (int j=tid;j<N_NODES;j+=nt) deg[j]=1;
  __syncthreads();
  for (int i=tid;i<L_DIM-1;i+=nt) atomicAdd(&deg[sk[i+1]],1);
  __syncthreads();
  for (int j=tid;j<N_NODES;j+=nt) dinv[j]=rsqrtf((float)deg[j]);
  for (int o=tid;o<N_NODES*GHD;o+=nt){
    int j=o>>3, c=o&7; float s=0.f;
    for (int d=0;d<DD;++d) s += k_emb[j*DD+d]*Wg1[d*GHD+c];
    h1[o]=s;
  }
  __syncthreads();
  for (int o=tid;o<N_NODES*GHD;o+=nt){ int j=o>>3; acc1[o]=h1[o]*dinv[j]*dinv[j]; }
  __syncthreads();
  for (int i=tid;i<L_DIM-1;i+=nt){
    int s=sk[i], d2=sk[i+1]; float coef=dinv[s]*dinv[d2];
    for (int c=0;c<GHD;++c) atomicAdd(&acc1[d2*GHD+c], h1[s*GHD+c]*coef);
  }
  __syncthreads();
  for (int o=tid;o<N_NODES*GHD;o+=nt){ int c=o&7; gact[o]=fmaxf(acc1[o]+bg1[c],0.f); }
  __syncthreads();
  for (int o=tid;o<N_NODES*DD;o+=nt){
    int j=o>>6, d=o&63; float s=0.f;
    for (int c=0;c<GHD;++c) s += gact[j*GHD+c]*Wg2[c*DD+d];
    gw[o]=s;
  }
  __syncthreads();
  for (int o=tid;o<N_NODES*DD;o+=nt){ int j=o>>6; acc2[o]=gw[o]*dinv[j]*dinv[j]; }
  __syncthreads();
  for (int i=tid;i<L_DIM-1;i+=nt){
    int s=sk[i], d2=sk[i+1]; float coef=dinv[s]*dinv[d2];
    for (int d=0;d<DD;++d) atomicAdd(&acc2[d2*DD+d], gw[s*DD+d]*coef);
  }
  __syncthreads();
  for (int o=tid;o<N_NODES*DD;o+=nt){ int d=o&63; stu[b*N_NODES*DD + o] = acc2[o]+bg2[d]; }
}

// ---------------------------------------------------------------- stu_vec (masked softmax pooling)
__global__ void __launch_bounds__(512)
stuvec_kernel(const int* __restrict__ skill, const int* __restrict__ answer,
              const float* __restrict__ pos, const float* __restrict__ stu,
              float* __restrict__ stu_vec)
{
  __shared__ float sprob[L_DIM];
  __shared__ float sred[512];
  __shared__ int   slen;
  __shared__ float spart[8*DD];
  const int b = blockIdx.x, tid = threadIdx.x;
  const int* ans = answer + b*L_DIM;
  const int* sk  = skill  + b*L_DIM;
  if (tid==0) slen=0;
  __syncthreads();
  int cnt=0;
  for (int l=tid;l<L_DIM;l+=512) cnt += (ans[l]!=2);
  atomicAdd(&slen, cnt);
  __syncthreads();
  int len = slen;
  int pidx = (len>0) ? (len-1) : (L_DIM-1);   // jnp negative-index wrap
  const float* prow = pos + pidx*L_DIM;
  for (int l=tid;l<L_DIM;l+=512){
    float m = (ans[l]!=2) ? 1.f : 0.f;
    sprob[l] = prow[l]*m;
  }
  __syncthreads();
  float mx=-1e30f;
  for (int l=tid;l<L_DIM;l+=512) mx=fmaxf(mx,sprob[l]);
  sred[tid]=mx; __syncthreads();
  for (int s=256;s>0;s>>=1){ if (tid<s) sred[tid]=fmaxf(sred[tid],sred[tid+s]); __syncthreads(); }
  mx=sred[0]; __syncthreads();
  float sm=0.f;
  for (int l=tid;l<L_DIM;l+=512){ float e=__expf(sprob[l]-mx); sprob[l]=e; sm+=e; }
  sred[tid]=sm; __syncthreads();
  for (int s=256;s>0;s>>=1){ if (tid<s) sred[tid]+=sred[tid+s]; __syncthreads(); }
  sm=sred[0]; __syncthreads();
  float inv=fastrcp_(sm);
  for (int l=tid;l<L_DIM;l+=512){
    float m=(ans[l]!=2)?1.f:0.f;
    sprob[l]=sprob[l]*inv*m;
  }
  __syncthreads();
  int g=tid>>6, d=tid&63;
  float acc=0.f;
  for (int l=g;l<L_DIM;l+=8) acc += sprob[l]*stu[(b*N_NODES + sk[l])*DD + d];
  spart[tid]=acc;
  __syncthreads();
  if (tid<DD){
    float s=0.f;
    for (int gg=0;gg<8;++gg) s += spart[gg*DD+tid];
    stu_vec[b*DD+tid]=s;
  }
}

// ---------------------------------------------------------------- gates -> k_h, v_h (f16)
__global__ void __launch_bounds__(64)
gates_kernel(const int* __restrict__ skill, const int* __restrict__ answer,
             const float* __restrict__ k_emb, const float* __restrict__ v_emb,
             const float* __restrict__ stu_vec,
             const float* __restrict__ Wgk, const float* __restrict__ bgk,
             const float* __restrict__ Wgv, const float* __restrict__ bgv,
             _Float16* __restrict__ k_h, _Float16* __restrict__ v_h)
{
  __shared__ float red[DD];
  __shared__ float gate[2];
  const int row = blockIdx.x;
  const int b   = row / L_DIM;
  const int d   = threadIdx.x;
  const int s   = skill[row];
  const int a   = answer[row];
  const int ax  = (a==2) ? 1 : a;
  const int xi  = s + NUM_C*ax;
  const float sv = stu_vec[b*DD+d];
  const float kd = k_emb[s*DD+d];
  const float vd = v_emb[xi*DD+d];
  red[d] = sv*Wgk[d] + kd*Wgk[DD+d];
  __syncthreads();
  if (d==0){ float t=0.f; for (int i=0;i<DD;++i) t+=red[i]; gate[0]=sigmoidf_(t+bgk[0]); }
  __syncthreads();
  float gk = gate[0];
  float kout = gk*sv + (1.f-gk)*kd;
  __syncthreads();
  red[d] = sv*Wgv[d] + vd*Wgv[DD+d];
  __syncthreads();
  if (d==0){ float t=0.f; for (int i=0;i<DD;++i) t+=red[i]; gate[1]=sigmoidf_(t+bgv[0]); }
  __syncthreads();
  float gv = gate[1];
  float vout = gv*sv + (1.f-gv)*vd;
  k_h[row*DD+d] = (_Float16)kout;
  v_h[row*DD+d] = (_Float16)vout;
}

// ---------------------------------------------------------------- WMMA: w=softmax(k@Mk^T), e, a
__global__ void __launch_bounds__(128)
gemm_wea_kernel(const _Float16* __restrict__ k_h, const _Float16* __restrict__ v_h,
                const _Float16* __restrict__ sw_all,   // [Mk|We|Wa] swizzled, 3*4096 halves
                const float* __restrict__ be, const float* __restrict__ ba,
                float* __restrict__ w_g, float* __restrict__ e_g, float* __restrict__ a_g)
{
  __shared__ _Float16 sW[3*4096];
  __shared__ float    sWt[4*16*DD];
  const int tid = threadIdx.x;
  stage_lds(sW, sw_all, 3*4096, tid, 128);
  __syncthreads();
  const _Float16* sMk = sW;
  const _Float16* sWe = sW + 4096;
  const _Float16* sWa = sW + 8192;
  const int wv = tid>>5, lane = tid&31;
  const int n = lane&15, hi = lane>>4;
  const int row0 = blockIdx.x*64 + wv*16;

  v16h ak0 = load_a_g(k_h, row0, lane, 0);
  v16h ak1 = load_a_g(k_h, row0, lane, 32);
  v16h av0 = load_a_g(v_h, row0, lane, 0);
  v16h av1 = load_a_g(v_h, row0, lane, 32);
  float* wt = sWt + wv*16*DD;

  #pragma unroll
  for (int nt2=0; nt2<4; ++nt2){
    v8f accW = {}; v8f accE = {}; v8f accA = {};
    accW = wmma_f16(ak0, load_b_sw(sMk, nt2*2+0, lane), accW);
    accW = wmma_f16(ak1, load_b_sw(sMk, nt2*2+1, lane), accW);
    accE = wmma_f16(av0, load_b_sw(sWe, nt2*2+0, lane), accE);
    accE = wmma_f16(av1, load_b_sw(sWe, nt2*2+1, lane), accE);
    accA = wmma_f16(av0, load_b_sw(sWa, nt2*2+0, lane), accA);
    accA = wmma_f16(av1, load_b_sw(sWa, nt2*2+1, lane), accA);
    #pragma unroll
    for (int r=0;r<8;++r){
      int mrow = r + hi*8;
      int grow = row0 + mrow;
      int col  = nt2*16 + n;
      wt[mrow*DD + col] = accW[r];
      e_g[grow*DD+col] = sigmoidf_ (accE[r] + be[col]);
      a_g[grow*DD+col] = fasttanh_(accA[r] + ba[col]);
    }
  }
  __syncthreads();
  if (lane < 16){                       // per-row softmax over 50 memory slots
    int mrow = lane;
    int grow = row0 + mrow;
    float mx=-1e30f;
    for (int c=0;c<M_DIM;++c) mx=fmaxf(mx, wt[mrow*DD+c]);
    float sm=0.f;
    for (int c=0;c<M_DIM;++c){ float e=__expf(wt[mrow*DD+c]-mx); wt[mrow*DD+c]=e; sm+=e; }
    float inv=fastrcp_(sm);
    for (int c=0;c<M_DIM;++c) w_g[grow*M_DIM+c] = wt[mrow*DD+c]*inv;
  }
}

// ---------------------------------------------------------------- sequential memory scan
__global__ void __launch_bounds__(256)
scan_kernel(const float* __restrict__ Mv0, const float* __restrict__ w_g,
            const float* __restrict__ e_g, const float* __restrict__ a_g,
            _Float16* __restrict__ r_h)
{
  __shared__ float Mv[M_DIM*DD];
  __shared__ float wv[M_DIM];
  __shared__ float ev[DD];
  __shared__ float avs[DD];
  __shared__ float part[4*DD];
  const int b = blockIdx.x, tid = threadIdx.x;
  for (int i=tid;i<M_DIM*DD;i+=256) Mv[i]=Mv0[i];
  __syncthreads();
  for (int t=0;t<L_DIM;++t){
    const int row = b*L_DIM + t;
    if (tid < M_DIM)                 wv[tid]      = w_g[row*M_DIM+tid];
    else if (tid>=64 && tid<128)     ev[tid-64]   = e_g[row*DD + tid-64];
    else if (tid>=128 && tid<192)    avs[tid-128] = a_g[row*DD + tid-128];
    __syncthreads();
    int g = tid>>6, d = tid&63;
    float p=0.f;
    for (int m=g;m<M_DIM;m+=4) p += wv[m]*Mv[m*DD+d];
    part[tid]=p;
    __syncthreads();
    if (tid < DD){
      float r = part[tid]+part[DD+tid]+part[2*DD+tid]+part[3*DD+tid];
      r_h[row*DD+tid] = (_Float16)r;
    }
    for (int i=tid;i<M_DIM*DD;i+=256){
      int m=i>>6, d2=i&63;
      Mv[i] = Mv[i]*(1.f - wv[m]*ev[d2]) + wv[m]*avs[d2];
    }
    __syncthreads();
  }
}

// ---------------------------------------------------------------- WMMA: f = tanh([r,k]@Wf + bf)
__global__ void __launch_bounds__(128)
gemm_f_kernel(const _Float16* __restrict__ r_h, const _Float16* __restrict__ k_h,
              const _Float16* __restrict__ swWf,   // 128x64 swizzled, 8192 halves
              const float* __restrict__ bf, float* __restrict__ f_g)
{
  __shared__ _Float16 sWf[8192];
  const int tid = threadIdx.x;
  stage_lds(sWf, swWf, 8192, tid, 128);
  __syncthreads();
  const int wv = tid>>5, lane = tid&31;
  const int n = lane&15, hi = lane>>4;
  const int row0 = blockIdx.x*64 + wv*16;

  v16h ar0 = load_a_g(r_h, row0, lane, 0);
  v16h ar1 = load_a_g(r_h, row0, lane, 32);
  v16h ak0 = load_a_g(k_h, row0, lane, 0);
  v16h ak1 = load_a_g(k_h, row0, lane, 32);

  #pragma unroll
  for (int nt2=0; nt2<4; ++nt2){
    v8f acc = {};
    acc = wmma_f16(ar0, load_b_sw(sWf, nt2*4+0, lane), acc);   // K 0..31  (r)
    acc = wmma_f16(ar1, load_b_sw(sWf, nt2*4+1, lane), acc);   // K 32..63 (r)
    acc = wmma_f16(ak0, load_b_sw(sWf, nt2*4+2, lane), acc);   // K 64..95 (k)
    acc = wmma_f16(ak1, load_b_sw(sWf, nt2*4+3, lane), acc);   // K 96..127(k)
    #pragma unroll
    for (int r=0;r<8;++r){
      int grow = row0 + r + hi*8;
      int col  = nt2*16 + n;
      f_g[grow*DD+col] = fasttanh_(acc[r] + bf[col]);
    }
  }
}

// ---------------------------------------------------------------- pred: single Wp column per (b,l)
__global__ void pred_kernel(const int* __restrict__ skill, const float* __restrict__ f_g,
                            const float* __restrict__ Wp, const float* __restrict__ bp,
                            float* __restrict__ out)
{
  int idx = blockIdx.x*blockDim.x + threadIdx.x;
  if (idx >= B_DIM*(L_DIM-1)) return;
  int b = idx/(L_DIM-1), l = idx%(L_DIM-1);
  int ns = skill[b*L_DIM + l + 1];
  const float* fr = f_g + (b*L_DIM + l)*DD;
  float acc=0.f;
  for (int d=0;d<DD;++d) acc += fr[d]*Wp[d*NUM_C+ns];
  out[idx] = sigmoidf_(acc + bp[ns]);
}

// ----------------------------------------------------------------
extern "C" void kernel_launch(void* const* d_in, const int* in_sizes, int n_in,
                              void* d_out, int out_size, void* d_ws, size_t ws_size,
                              hipStream_t stream) {
  const int*   skill  = (const int*)  d_in[0];
  const int*   answer = (const int*)  d_in[1];
  const float* k_emb  = (const float*)d_in[2];
  const float* v_emb  = (const float*)d_in[3];
  const float* Mk     = (const float*)d_in[4];
  const float* Mv0    = (const float*)d_in[5];
  const float* pos    = (const float*)d_in[6];
  const float* Wgk    = (const float*)d_in[7];
  const float* bgk    = (const float*)d_in[8];
  const float* Wgv    = (const float*)d_in[9];
  const float* bgv    = (const float*)d_in[10];
  const float* Wf     = (const float*)d_in[11];
  const float* bf     = (const float*)d_in[12];
  const float* Wp     = (const float*)d_in[13];
  const float* bp     = (const float*)d_in[14];
  const float* We     = (const float*)d_in[15];
  const float* be     = (const float*)d_in[16];
  const float* Wa     = (const float*)d_in[17];
  const float* ba     = (const float*)d_in[18];
  const float* Wg1    = (const float*)d_in[19];
  const float* bg1    = (const float*)d_in[20];
  const float* Wg2    = (const float*)d_in[21];
  const float* bg2    = (const float*)d_in[22];
  float* out = (float*)d_out;

  char* ws = (char*)d_ws;
  auto carve = [&](size_t bytes)->char*{
    char* p = ws;
    ws += (bytes + 255) & ~(size_t)255;
    return p;
  };
  const size_t BL = (size_t)B_DIM*L_DIM;
  float*    stu     = (float*)   carve((size_t)B_DIM*N_NODES*DD*sizeof(float));
  float*    stu_vec = (float*)   carve((size_t)B_DIM*DD*sizeof(float));
  _Float16* k_h     = (_Float16*)carve(BL*DD*sizeof(_Float16));
  _Float16* v_h     = (_Float16*)carve(BL*DD*sizeof(_Float16));
  float*    w_g     = (float*)   carve(BL*M_DIM*sizeof(float));
  float*    e_g     = (float*)   carve(BL*DD*sizeof(float));
  float*    a_g     = (float*)   carve(BL*DD*sizeof(float));
  _Float16* r_h     = (_Float16*)carve(BL*DD*sizeof(_Float16));
  float*    f_g     = (float*)   carve(BL*DD*sizeof(float));
  _Float16* sw_all  = (_Float16*)carve((size_t)3*4096*sizeof(_Float16)); // [Mk|We|Wa]
  _Float16* sw_wf   = (_Float16*)carve((size_t)8192*sizeof(_Float16));
  (void)ws_size; (void)in_sizes; (void)n_in; (void)out_size;

  swizzle_weights_kernel<<<32, 256, 0, stream>>>(Mk, We, Wa, Wf,
                                                 sw_all, sw_all+4096, sw_all+8192, sw_wf);
  gcn_kernel     <<<B_DIM, 128, 0, stream>>>(skill, k_emb, Wg1, bg1, Wg2, bg2, stu);
  stuvec_kernel  <<<B_DIM, 512, 0, stream>>>(skill, answer, pos, stu, stu_vec);
  gates_kernel   <<<(int)BL, 64, 0, stream>>>(skill, answer, k_emb, v_emb, stu_vec,
                                              Wgk, bgk, Wgv, bgv, k_h, v_h);
  gemm_wea_kernel<<<(int)(BL/64), 128, 0, stream>>>(k_h, v_h, sw_all, be, ba,
                                                    w_g, e_g, a_g);
  scan_kernel    <<<B_DIM, 256, 0, stream>>>(Mv0, w_g, e_g, a_g, r_h);
  gemm_f_kernel  <<<(int)(BL/64), 128, 0, stream>>>(r_h, k_h, sw_wf, bf, f_g);
  int npred = B_DIM*(L_DIM-1);
  pred_kernel    <<<(npred+255)/256, 256, 0, stream>>>(skill, f_g, Wp, bp, out);
}
